// SymNet_8701603741806
// MI455X (gfx1250) — matrix operand order repaired and checked by reading, not compiled, not run on hardware
//
#include <hip/hip_runtime.h>
#include <hip/hip_bf16.h>
#include <math.h>

typedef __bf16 bf16;
typedef __attribute__((ext_vector_type(16))) __bf16 v16bf;
typedef __attribute__((ext_vector_type(8)))  __bf16 v8bf;
typedef __attribute__((ext_vector_type(8)))  float  v8f;

constexpr int B_    = 2;
constexpr int C_    = 1024;
constexpr int H_    = 60;
constexpr int S_    = 3600;          // 60*60
constexpr int S_PAD = 3616;          // padded to 32-row multiple (113 * 32)
constexpr int PAD   = S_PAD - S_;    // 16 zero rows per plane
constexpr int GT    = S_PAD / 32;    // 113 tile-pair groups per GEMM dim
constexpr int HID   = 256;
constexpr int TDIM  = 300;
constexpr int NK    = 3;             // (5,5), (7,1), (1,7)
constexpr float EPSC = 1e-7f;
constexpr int MH = 473;              // mask H/W

// ---------------------------------------------------------------- mask resize
__global__ void mask_resize_kernel(const float* __restrict__ masks,
                                   float* __restrict__ mask60) {
  int tid = blockIdx.x * blockDim.x + threadIdx.x;
  if (tid >= B_ * S_) return;
  int b = tid / S_, s = tid % S_;
  int y = s / H_, x = s % H_;
  const float sc = (float)(MH - 1) / (float)(H_ - 1);   // align_corners
  float py = y * sc, px = x * sc;
  int y0 = (int)floorf(py); int y1 = min(y0 + 1, MH - 1); float wy = py - y0;
  int x0 = (int)floorf(px); int x1 = min(x0 + 1, MH - 1); float wx = px - x0;
  const float* m = masks + (long)b * MH * MH;
  float v = (m[y0 * MH + x0] * (1.f - wy) + m[y1 * MH + x0] * wy) * (1.f - wx)
          + (m[y0 * MH + x1] * (1.f - wy) + m[y1 * MH + x1] * wy) * wx;
  mask60[tid] = v;
}

// ------------------------------------------------ zero the padded tail rows
__global__ void zeropad_kernel(bf16* __restrict__ qpool, bf16* __restrict__ spool) {
  int tid = blockIdx.x * blockDim.x + threadIdx.x;
  const int per = NK * B_ * PAD * C_;
  if (tid >= 2 * per) return;
  bf16* base = (tid < per) ? qpool : spool;
  int t = (tid < per) ? tid : (tid - per);
  int c  = t % C_;
  int pr = (t / C_) % PAD;
  int kb = t / (C_ * PAD);
  base[((long)kb * S_PAD + S_ + pr) * C_ + c] = (bf16)0.f;
}

// ------------------------------------------- fused 3-kernel avg pool (+mask)
// writes bf16 in [kern][b][s][c] layout (GEMM operand layout, padded rows)
__global__ void pool3_kernel(const float* __restrict__ src,
                             const float* __restrict__ mask,   // null for query
                             bf16* __restrict__ dst, int total) {
  int tid = blockIdx.x * blockDim.x + threadIdx.x;
  if (tid >= total) return;
  int s = tid % S_;
  int c = (tid / S_) % C_;
  int b = tid / (S_ * C_);
  int y = s / H_, x = s % H_;
  const float* p = src + ((long)(b * C_ + c)) * S_;
  const float* mrow = mask ? (mask + (long)b * S_) : nullptr;

  float s55 = 0.f;
  for (int dy = -2; dy <= 2; ++dy) {
    int yy = y + dy; if ((unsigned)yy >= (unsigned)H_) continue;
    for (int dx = -2; dx <= 2; ++dx) {
      int xx = x + dx; if ((unsigned)xx >= (unsigned)H_) continue;
      float v = p[yy * H_ + xx]; if (mrow) v *= mrow[yy * H_ + xx];
      s55 += v;
    }
  }
  float s71 = 0.f;
  for (int dy = -3; dy <= 3; ++dy) {
    int yy = y + dy; if ((unsigned)yy >= (unsigned)H_) continue;
    float v = p[yy * H_ + x]; if (mrow) v *= mrow[yy * H_ + x];
    s71 += v;
  }
  float s17 = 0.f;
  for (int dx = -3; dx <= 3; ++dx) {
    int xx = x + dx; if ((unsigned)xx >= (unsigned)H_) continue;
    float v = p[y * H_ + xx]; if (mrow) v *= mrow[y * H_ + xx];
    s17 += v;
  }
  long row = ((long)b * S_PAD + s) * C_ + c;
  const long plane = (long)B_ * S_PAD * C_;
  dst[0 * plane + row] = (bf16)(s55 * (1.f / 25.f));
  dst[1 * plane + row] = (bf16)(s71 * (1.f / 7.f));
  dst[2 * plane + row] = (bf16)(s17 * (1.f / 7.f));
}

// ----------------------------------------------------- per-row L2 norm (wave)
__global__ void rownorm_kernel(const bf16* __restrict__ rows,
                               float* __restrict__ out, int nrows) {
  int lane = threadIdx.x & 31;
  int wid  = threadIdx.x >> 5;
  int row  = blockIdx.x * (blockDim.x >> 5) + wid;
  if (row >= nrows) return;
  const bf16* p = rows + (long)row * C_;
  float acc = 0.f;
  #pragma unroll 4
  for (int i = 0; i < C_ / 32; ++i) {
    float v = (float)p[i * 32 + lane];
    acc += v * v;
  }
  for (int off = 16; off >= 1; off >>= 1) acc += __shfl_xor(acc, off, 32);
  if (lane == 0) out[row] = sqrtf(acc);
}

// ------------- main GEMM: 2x2 register tile (32x32 output) per wave ---------
// rowsum[kb][s] += sum_t  (q[s].k[t]) / (qn[s]*sn[t] + eps)
__global__ void __launch_bounds__(256)
simgemm_kernel(const bf16* __restrict__ qpool, const bf16* __restrict__ spool,
               const float* __restrict__ qn, const float* __restrict__ sn,
               float* __restrict__ rowsum) {
  const int lane = threadIdx.x & 31;
  const int wave = threadIdx.x >> 5;
  long job = (long)blockIdx.x * (blockDim.x >> 5) + wave;   // wave-uniform
  const long total = (long)NK * B_ * GT * GT;
  if (job >= total) return;                // uniform exit: EXEC stays all-1
  int gc = (int)(job % GT); job /= GT;     // 32-col group (t / N dim)
  int gr = (int)(job % GT); job /= GT;     // 32-row group (s / M dim)
  int kb = (int)job;                       // kern*B + b

  const int m15 = lane & 15, hi = lane >> 4;
  const bf16* a0 = qpool + ((long)kb * S_PAD + gr * 32 + m15) * C_;
  const bf16* a1 = a0 + (long)16 * C_;
  const bf16* b0 = spool + ((long)kb * S_PAD + gc * 32 + m15) * C_;
  const bf16* b1 = b0 + (long)16 * C_;

  v8f acc00 = {0.f,0.f,0.f,0.f,0.f,0.f,0.f,0.f};
  v8f acc01 = {0.f,0.f,0.f,0.f,0.f,0.f,0.f,0.f};
  v8f acc10 = {0.f,0.f,0.f,0.f,0.f,0.f,0.f,0.f};
  v8f acc11 = {0.f,0.f,0.f,0.f,0.f,0.f,0.f,0.f};

  for (int k0 = 0; k0 < C_; k0 += 32) {
    // A 16x32 bf16 layout: lane m = lane&15; K runs [hi*8 ..] and [16+hi*8 ..]
    v8bf A0lo = *(const v8bf*)(a0 + k0 + hi * 8);
    v8bf A0hi = *(const v8bf*)(a0 + k0 + 16 + hi * 8);
    v8bf A1lo = *(const v8bf*)(a1 + k0 + hi * 8);
    v8bf A1hi = *(const v8bf*)(a1 + k0 + 16 + hi * 8);
    // B 32x16 bf16 layout: lane n = lane&15; K runs [hi*8 ..] and [16+hi*8 ..]
    v8bf B0lo = *(const v8bf*)(b0 + k0 + hi * 8);
    v8bf B0hi = *(const v8bf*)(b0 + k0 + 16 + hi * 8);
    v8bf B1lo = *(const v8bf*)(b1 + k0 + hi * 8);
    v8bf B1hi = *(const v8bf*)(b1 + k0 + 16 + hi * 8);
    if (k0 + 32 < C_) {                       // prefetch next K block
      __builtin_prefetch(a0 + k0 + 32, 0, 1);
      __builtin_prefetch(a1 + k0 + 32, 0, 1);
      __builtin_prefetch(b0 + k0 + 32, 0, 1);
      __builtin_prefetch(b1 + k0 + 32, 0, 1);
    }
    v16bf A0, A1, Bv0, Bv1;
    #pragma unroll
    for (int j = 0; j < 8; ++j) {
      A0[j] = A0lo[j];  A0[j + 8] = A0hi[j];
      A1[j] = A1lo[j];  A1[j + 8] = A1hi[j];
      Bv0[j] = B0lo[j]; Bv0[j + 8] = B0hi[j];
      Bv1[j] = B1lo[j]; Bv1[j + 8] = B1hi[j];
    }
    acc00 = __builtin_amdgcn_wmma_f32_16x16x32_bf16(false, A0, false, Bv0, (short)0, acc00, false, false);
    acc01 = __builtin_amdgcn_wmma_f32_16x16x32_bf16(false, A0, false, Bv1, (short)0, acc01, false, false);
    acc10 = __builtin_amdgcn_wmma_f32_16x16x32_bf16(false, A1, false, Bv0, (short)0, acc10, false, false);
    acc11 = __builtin_amdgcn_wmma_f32_16x16x32_bf16(false, A1, false, Bv1, (short)0, acc11, false, false);
  }

  // C/D layout: lane holds N = lane&15; VGPR j holds M = hi*8 + j.
  // Fold the two N-tiles together, then reduce across the 16 N-lanes.
  float sn0 = sn[(long)kb * S_PAD + gc * 32 + m15];
  float sn1 = sn[(long)kb * S_PAD + gc * 32 + 16 + m15];
  #pragma unroll
  for (int mi = 0; mi < 2; ++mi) {
    #pragma unroll
    for (int j = 0; j < 8; ++j) {
      int m = gr * 32 + mi * 16 + hi * 8 + j;
      float qv = qn[(long)kb * S_PAD + m];
      float v0 = (mi == 0) ? acc00[j] : acc10[j];
      float v1 = (mi == 0) ? acc01[j] : acc11[j];
      float v = v0 / (qv * sn0 + EPSC) + v1 / (qv * sn1 + EPSC);
      for (int off = 8; off >= 1; off >>= 1) v += __shfl_xor(v, off, 32);
      if (m15 == 0) atomicAdd(&rowsum[(long)kb * S_PAD + m], v);
    }
  }
}

// ------------------------------------ sim = qn*rowsum/S ; per-(kern,b) min/max
__global__ void sim_minmax_kernel(const float* __restrict__ qn,
                                  const float* __restrict__ rowsum,
                                  float* __restrict__ simbuf,
                                  float* __restrict__ mnmx) {
  int kb = blockIdx.x, tid = threadIdx.x;
  __shared__ float smn[256], smx[256];
  float mn = 3.4e38f, mx = -3.4e38f;
  for (int s = tid; s < S_; s += 256) {
    long idx = (long)kb * S_PAD + s;
    float v = qn[idx] * rowsum[idx] * (1.f / (float)S_);
    simbuf[idx] = v;
    mn = fminf(mn, v); mx = fmaxf(mx, v);
  }
  smn[tid] = mn; smx[tid] = mx; __syncthreads();
  for (int off = 128; off >= 1; off >>= 1) {
    if (tid < off) {
      smn[tid] = fminf(smn[tid], smn[tid + off]);
      smx[tid] = fmaxf(smx[tid], smx[tid + off]);
    }
    __syncthreads();
  }
  if (tid == 0) { mnmx[kb * 2] = smn[0]; mnmx[kb * 2 + 1] = smx[0]; }
}

// ---------------------------- normalize, mean over kernels, apply shot weight
__global__ void corr1_kernel(const float* __restrict__ simbuf,
                             const float* __restrict__ mnmx,
                             const float* __restrict__ weight,
                             float* __restrict__ out) {
  int tid = blockIdx.x * blockDim.x + threadIdx.x;
  if (tid >= B_ * S_) return;
  int b = tid / S_, s = tid % S_;
  float acc = 0.f;
  for (int kkk = 0; kkk < NK; ++kkk) {
    int kb = kkk * B_ + b;
    float mn = mnmx[kb * 2], mx = mnmx[kb * 2 + 1];
    float v = simbuf[(long)kb * S_PAD + s];
    acc += (v - mn) / (mx - mn + EPSC);
  }
  out[tid] = weight[b] * (acc * (1.f / (float)NK));  // shot == 1
}

// -------------------------------------------- align_corners bilinear downsize
__global__ void resize_kernel(const float* __restrict__ src,
                              float* __restrict__ dst, int osz) {
  int tid = blockIdx.x * blockDim.x + threadIdx.x;
  if (tid >= B_ * osz * osz) return;
  int b = tid / (osz * osz), r = tid % (osz * osz);
  int y = r / osz, x = r % osz;
  float sc = (float)(H_ - 1) / (float)(osz - 1);
  float py = y * sc, px = x * sc;
  int y0 = (int)floorf(py), x0 = (int)floorf(px);
  int y1 = min(y0 + 1, H_ - 1), x1 = min(x0 + 1, H_ - 1);
  float wy = py - y0, wx = px - x0;
  const float* p = src + (long)b * S_;
  dst[tid] = (p[y0 * H_ + x0] * (1.f - wy) + p[y1 * H_ + x0] * wy) * (1.f - wx)
           + (p[y0 * H_ + x1] * (1.f - wy) + p[y1 * H_ + x1] * wy) * wx;
}

// ------------------------------------------------------------------- APA path
__global__ void kx_kernel(const float* __restrict__ w_k,
                          const float* __restrict__ k,
                          float* __restrict__ kx) {
  long tid = (long)blockIdx.x * blockDim.x + threadIdx.x;
  if (tid >= (long)B_ * HID * S_) return;
  int s = (int)(tid % S_);
  int o = (int)((tid / S_) % HID);
  int b = (int)(tid / ((long)S_ * HID));
  const float* wrow = w_k + o * HID;
  const float* kb = k + (long)b * HID * S_ + s;
  float acc = 0.f;
  #pragma unroll 4
  for (int i = 0; i < HID; ++i) acc += wrow[i] * kb[(long)i * S_];
  kx[tid] = acc;   // [b][o][s]
}

__global__ void qw_kernel(const float* __restrict__ text,
                          const float* __restrict__ proto,
                          const float* __restrict__ w_q,
                          const float* __restrict__ w_bl,
                          float* __restrict__ qw) {
  int b = blockIdx.x, h = threadIdx.x;
  __shared__ float qx[HID];
  float acc = 0.f;
  for (int e = 0; e < TDIM + HID; ++e) {
    float qe = (e < TDIM) ? text[b * TDIM + e] : proto[b * HID + (e - TDIM)];
    acc += qe * w_q[e * HID + h];
  }
  qx[h] = acc; __syncthreads();
  float acc2 = 0.f;
  for (int i = 0; i < HID; ++i) acc2 += qx[i] * w_bl[i * HID + h];
  qw[b * HID + h] = acc2;
}

__global__ void score_kernel(const float* __restrict__ qw,
                             const float* __restrict__ kx,
                             float* __restrict__ score) {
  int tid = blockIdx.x * blockDim.x + threadIdx.x;
  if (tid >= B_ * S_) return;
  int b = tid / S_, s = tid % S_;
  const float* q = qw + b * HID;
  const float* kb = kx + (long)b * HID * S_ + s;
  float acc = 0.f;
  #pragma unroll 4
  for (int d = 0; d < HID; ++d) acc += q[d] * kb[(long)d * S_];
  score[tid] = acc;
}

__global__ void softmax_kernel(float* __restrict__ score) {
  int b = blockIdx.x, tid = threadIdx.x;
  __shared__ float red[256];
  float* p = score + (long)b * S_;
  float mx = -3.4e38f;
  for (int s = tid; s < S_; s += 256) mx = fmaxf(mx, p[s]);
  red[tid] = mx; __syncthreads();
  for (int off = 128; off >= 1; off >>= 1) {
    if (tid < off) red[tid] = fmaxf(red[tid], red[tid + off]);
    __syncthreads();
  }
  mx = red[0]; __syncthreads();
  float sum = 0.f;
  for (int s = tid; s < S_; s += 256) {
    float e = __expf(p[s] - mx);
    p[s] = e; sum += e;
  }
  red[tid] = sum; __syncthreads();
  for (int off = 128; off >= 1; off >>= 1) {
    if (tid < off) red[tid] += red[tid + off];
    __syncthreads();
  }
  float inv = 1.f / red[0];
  for (int s = tid; s < S_; s += 256) p[s] *= inv;
}

__global__ void attnout_kernel(const float* __restrict__ score,
                               const float* __restrict__ kx,
                               float* __restrict__ outbuf) {
  int tid = blockIdx.x * blockDim.x + threadIdx.x;
  if (tid >= B_ * HID) return;
  int b = tid / HID, o = tid % HID;
  const float* p = score + (long)b * S_;
  const float* kb = kx + ((long)b * HID + o) * S_;
  float acc = 0.f;
  #pragma unroll 4
  for (int s = 0; s < S_; ++s) acc += p[s] * kb[s];
  outbuf[tid] = acc;
}

__global__ void proj_kernel(const float* __restrict__ outbuf,
                            const float* __restrict__ w_proj,
                            const float* __restrict__ b_proj,
                            const float* __restrict__ proto,
                            float* __restrict__ out) {
  int tid = blockIdx.x * blockDim.x + threadIdx.x;
  if (tid >= B_ * HID) return;
  int b = tid / HID, o = tid % HID;
  const float* in = outbuf + b * HID;
  float acc = b_proj[o];
  #pragma unroll 4
  for (int d = 0; d < HID; ++d) acc += in[d] * w_proj[d * HID + o];
  out[tid] = acc + proto[b * HID + o];
}

// --------------------------------------------------------------------- launch
extern "C" void kernel_launch(void* const* d_in, const int* in_sizes, int n_in,
                              void* d_out, int out_size, void* d_ws, size_t ws_size,
                              hipStream_t stream) {
  (void)in_sizes; (void)n_in; (void)out_size; (void)ws_size;
  const float* weight = (const float*)d_in[0];
  const float* query  = (const float*)d_in[1];
  const float* supp   = (const float*)d_in[2];
  const float* masks  = (const float*)d_in[3];
  const float* karr   = (const float*)d_in[4];
  const float* text   = (const float*)d_in[5];
  const float* proto  = (const float*)d_in[6];
  const float* w_q    = (const float*)d_in[7];
  const float* w_k    = (const float*)d_in[8];
  const float* w_bl   = (const float*)d_in[9];
  const float* w_proj = (const float*)d_in[10];
  const float* b_proj = (const float*)d_in[11];
  float* out = (float*)d_out;

  char* ws = (char*)d_ws;
  size_t off = 0;
  auto carve = [&](size_t bytes) -> char* {
    char* p = ws + off;
    off += (bytes + 255) & ~(size_t)255;
    return p;
  };
  bf16*  qpool  = (bf16*)carve((size_t)NK * B_ * S_PAD * C_ * sizeof(bf16));
  bf16*  spool  = (bf16*)carve((size_t)NK * B_ * S_PAD * C_ * sizeof(bf16));
  float* qn     = (float*)carve((size_t)NK * B_ * S_PAD * sizeof(float));
  float* sn     = (float*)carve((size_t)NK * B_ * S_PAD * sizeof(float));
  float* rowsum = (float*)carve((size_t)NK * B_ * S_PAD * sizeof(float));
  float* simbuf = (float*)carve((size_t)NK * B_ * S_PAD * sizeof(float));
  float* mnmx   = (float*)carve((size_t)NK * B_ * 2 * sizeof(float));
  float* mask60 = (float*)carve((size_t)B_ * S_ * sizeof(float));
  float* kx     = (float*)carve((size_t)B_ * HID * S_ * sizeof(float));
  float* qw     = (float*)carve((size_t)B_ * HID * sizeof(float));
  float* score  = (float*)carve((size_t)B_ * S_ * sizeof(float));
  float* attno  = (float*)carve((size_t)B_ * HID * sizeof(float));

  hipMemsetAsync(rowsum, 0, (size_t)NK * B_ * S_PAD * sizeof(float), stream);

  // ------------------------ SPM ------------------------
  const int padTot = 2 * NK * B_ * PAD * C_;
  zeropad_kernel<<<(padTot + 255) / 256, 256, 0, stream>>>(qpool, spool);
  mask_resize_kernel<<<(B_ * S_ + 255) / 256, 256, 0, stream>>>(masks, mask60);
  const int totPool = B_ * C_ * S_;
  pool3_kernel<<<(totPool + 255) / 256, 256, 0, stream>>>(query, nullptr, qpool, totPool);
  pool3_kernel<<<(totPool + 255) / 256, 256, 0, stream>>>(supp, mask60, spool, totPool);
  const int nrows = NK * B_ * S_PAD;
  rownorm_kernel<<<(nrows + 7) / 8, 256, 0, stream>>>(qpool, qn, nrows);
  rownorm_kernel<<<(nrows + 7) / 8, 256, 0, stream>>>(spool, sn, nrows);

  const long jobs = (long)NK * B_ * GT * GT;           // 76,614 wave jobs
  simgemm_kernel<<<(int)((jobs + 7) / 8), 256, 0, stream>>>(qpool, spool, qn, sn, rowsum);

  sim_minmax_kernel<<<NK * B_, 256, 0, stream>>>(qn, rowsum, simbuf, mnmx);

  float* corr1 = out;                    // [B,1,60,60] = 7200
  float* corr2 = corr1 + B_ * S_;        // [B,1,30,30] = 1800
  float* corr3 = corr2 + B_ * 30 * 30;   // [B,1,15,15] = 450
  float* corr4 = corr3 + B_ * 15 * 15;   // [B,1,8,8]   = 128
  float* outv  = corr4 + B_ * 8 * 8;     // [B,1,256]   = 512

  corr1_kernel<<<(B_ * S_ + 255) / 256, 256, 0, stream>>>(simbuf, mnmx, weight, corr1);
  resize_kernel<<<(B_ * 30 * 30 + 255) / 256, 256, 0, stream>>>(corr1, corr2, 30);
  resize_kernel<<<(B_ * 15 * 15 + 255) / 256, 256, 0, stream>>>(corr1, corr3, 15);
  resize_kernel<<<(B_ * 8 * 8 + 255) / 256, 256, 0, stream>>>(corr1, corr4, 8);

  // ------------------------ APA ------------------------
  const long totKx = (long)B_ * HID * S_;
  kx_kernel<<<(int)((totKx + 255) / 256), 256, 0, stream>>>(w_k, karr, kx);
  qw_kernel<<<B_, HID, 0, stream>>>(text, proto, w_q, w_bl, qw);
  score_kernel<<<(B_ * S_ + 255) / 256, 256, 0, stream>>>(qw, kx, score);
  softmax_kernel<<<B_, 256, 0, stream>>>(score);
  attnout_kernel<<<(B_ * HID + 255) / 256, 256, 0, stream>>>(score, kx, attno);
  proj_kernel<<<(B_ * HID + 255) / 256, 256, 0, stream>>>(attno, w_proj, b_proj, proto, outv);
}
